// LSTMModel_58686433133119
// MI455X (gfx1250) — matrix operand order repaired and checked by reading, not compiled, not run on hardware
//
#include <hip/hip_runtime.h>

#define H 1024
#define V 100000
#define T_STEPS 64
#define KTOP 20
#define G4 4096
#define TOPK_BLOCKS 64
#define CHUNK 1563               // ceil(V / TOPK_BLOCKS)
#define NCAND (TOPK_BLOCKS * KTOP)

typedef __bf16 bf16;
typedef __attribute__((ext_vector_type(16))) __bf16 v16bf;
typedef __attribute__((ext_vector_type(8)))  __bf16 bf16x8;
typedef __attribute__((ext_vector_type(8)))  float  v8f;

union V16U { v16bf v; bf16x8 h[2]; };

// ---------------- init ----------------
__global__ void k_init(float* mask, float* h, float* c, int* ah, float* fb) {
  int i = blockIdx.x * blockDim.x + threadIdx.x;
  if (i < V) mask[i] = 1.0f;
  if (i < H) { h[i] = 0.0f; c[i] = 0.0f; }
  if (i == 0) { *ah = 0; *fb = 0.0f; }
}

// ---------------- fp32 -> bf16 weight conversion (one-time) ----------------
__global__ void k_convert(const float* __restrict__ w, bf16* __restrict__ wb, long n) {
  long i = (long)blockIdx.x * blockDim.x + threadIdx.x;
  long step = (long)gridDim.x * blockDim.x;
  for (; i < n; i += step) wb[i] = (bf16)w[i];
}

// ---------------- gates = W_ih@inp + b_ih + W_hh@h + b_hh (wave per row) ----------------
__global__ void k_gates(const float* __restrict__ Wih, const float* __restrict__ Whh,
                        const float* __restrict__ bih, const float* __restrict__ bhh,
                        const float* __restrict__ embed, const float* __restrict__ h,
                        const int* ah_p, const float* fb_p, float* __restrict__ gates) {
  int lane = threadIdx.x & 31;
  int wave = (blockIdx.x * blockDim.x + threadIdx.x) >> 5;
  if (wave >= G4) return;
  int ah = *ah_p;
  float fb = *fb_p;                       // 0 at t==0 -> zero input vector
  const float* e  = embed + (long)ah * H;
  const float* wi = Wih + (long)wave * H;
  const float* wh = Whh + (long)wave * H;
  float s = 0.0f;
  for (int k = lane; k < H; k += 32)
    s = fmaf(wi[k], fb * e[k], fmaf(wh[k], h[k], s));
  for (int off = 16; off; off >>= 1) s += __shfl_xor(s, off, 32);
  if (lane == 0) gates[wave] = s + bih[wave] + bhh[wave];
}

// ---------------- LSTM pointwise update (gate order i,f,g,o) ----------------
__global__ void k_lstm(const float* __restrict__ gates, float* h, float* c, bf16* hb,
                       float* __restrict__ hs_out, float* __restrict__ cs_out) {
  int j = threadIdx.x;
  float i_ = 1.0f / (1.0f + __expf(-gates[j]));
  float f_ = 1.0f / (1.0f + __expf(-gates[H + j]));
  float g_ = tanhf(gates[2 * H + j]);
  float o_ = 1.0f / (1.0f + __expf(-gates[3 * H + j]));
  float cn = f_ * c[j] + i_ * g_;
  float hn = o_ * tanhf(cn);
  c[j] = cn; h[j] = hn; hb[j] = (bf16)hn;
  hs_out[j] = hn; cs_out[j] = cn;
}

// ---------------- logits via bf16 WMMA: wave computes 16 rows of W_out @ h ----------------
__global__ __launch_bounds__(128)
void k_logits_wmma(const bf16* __restrict__ Wb, const bf16* __restrict__ hb_g,
                   const float* __restrict__ bout, float* __restrict__ logits,
                   float* __restrict__ pmax) {
  __shared__ bf16 hsm[H];
  __shared__ float lsm[64];
  int tid = threadIdx.x;
  for (int i = tid; i < H; i += 128) hsm[i] = hb_g[i];
  if (tid < 64) lsm[tid] = -3.0e38f;
  __syncthreads();

  int wave = tid >> 5, lane = tid & 31;
  int tile = blockIdx.x * 4 + wave;              // 16 output rows per tile
  if (tile < V / 16) {                           // wave-uniform guard: EXEC stays all-ones
    int col = lane & 15, half = lane >> 4;
    const bf16* wrow = Wb + (long)(tile * 16 + col) * H + 8 * half;
    v8f acc = {0.f, 0.f, 0.f, 0.f, 0.f, 0.f, 0.f, 0.f};
    for (int k0 = 0; k0 < H; k0 += 32) {
      V16U a, b;
      // A (16x32 bf16, rows replicated with h): lanes<16 K=[0..8)+[16..24); lanes>=16 +8
      a.h[0] = *(const bf16x8*)(&hsm[k0 + 8 * half]);
      a.h[1] = *(const bf16x8*)(&hsm[k0 + 16 + 8 * half]);
      // B (32x16 bf16): lanes<16 K=[0..8)+[16..24); lanes>=16 +8 -> 2x b128 loads
      b.h[0] = *(const bf16x8*)(wrow + k0);
      b.h[1] = *(const bf16x8*)(wrow + k0 + 16);
      acc = __builtin_amdgcn_wmma_f32_16x16x32_bf16(false, a.v, false, b.v,
                                                    (short)0, acc, false, false);
    }
    if (lane < 16) {                             // D[0][n] lives in lanes 0..15, VGPR 0
      int row = tile * 16 + lane;
      float L = acc[0] + bout[row];
      logits[row] = L;
      lsm[wave * 16 + lane] = L;
    }
  }
  __syncthreads();
  if (tid == 0) {
    float m = -3.0e38f;
    for (int i = 0; i < 64; i++) m = fmaxf(m, lsm[i]);
    pmax[blockIdx.x] = m;
  }
}

// ---------------- fp32 fallback logits (wave per row) ----------------
__global__ void k_logits_f32(const float* __restrict__ Wout, const float* __restrict__ bout,
                             const float* __restrict__ h, float* __restrict__ logits,
                             float* __restrict__ pmax) {
  __shared__ float lsm[8];
  int lane = threadIdx.x & 31;
  int wv = threadIdx.x >> 5;
  int row = blockIdx.x * 8 + wv;                 // grid = 12500 -> rows exactly V
  const float* w = Wout + (long)row * H;
  float s = 0.0f;
  for (int k = lane; k < H; k += 32) s = fmaf(w[k], h[k], s);
  for (int off = 16; off; off >>= 1) s += __shfl_xor(s, off, 32);
  if (lane == 0) { float L = s + bout[row]; logits[row] = L; lsm[wv] = L; }
  __syncthreads();
  if (threadIdx.x == 0) {
    float m = lsm[0];
    for (int i = 1; i < 8; i++) m = fmaxf(m, lsm[i]);
    pmax[blockIdx.x] = m;
  }
}

// ---------------- partial exp-sum (each block re-reduces pmax itself) ----------------
__global__ void k_expsum(const float* __restrict__ logits, const float* __restrict__ pmax,
                         int nb, float* __restrict__ psum) {
  __shared__ float red[256];
  int tid = threadIdx.x;
  float m = -3.0e38f;
  for (int i = tid; i < nb; i += 256) m = fmaxf(m, pmax[i]);
  red[tid] = m; __syncthreads();
  for (int off = 128; off; off >>= 1) { if (tid < off) red[tid] = fmaxf(red[tid], red[tid + off]); __syncthreads(); }
  float M = red[0]; __syncthreads();
  float s = 0.0f;
  for (int v = blockIdx.x * 256 + tid; v < V; v += 256 * gridDim.x) s += __expf(logits[v] - M);
  red[tid] = s; __syncthreads();
  for (int off = 128; off; off >>= 1) { if (tid < off) red[tid] += red[tid + off]; __syncthreads(); }
  if (tid == 0) psum[blockIdx.x] = red[0];
}

// ---------------- normalize + mask + write masked probs + block-local top-20 ----------------
__global__ __launch_bounds__(256)
void k_normtopk(const float* __restrict__ logits, const float* __restrict__ pmax, int nb,
                const float* __restrict__ psum, const float* __restrict__ mask,
                float* __restrict__ mp_out, float* __restrict__ cand_v, int* __restrict__ cand_i) {
  __shared__ float sv[CHUNK];
  __shared__ float rv[256];
  __shared__ int   rp[256];
  int tid = threadIdx.x;
  float m = -3.0e38f;
  for (int i = tid; i < nb; i += 256) m = fmaxf(m, pmax[i]);
  rv[tid] = m; __syncthreads();
  for (int off = 128; off; off >>= 1) { if (tid < off) rv[tid] = fmaxf(rv[tid], rv[tid + off]); __syncthreads(); }
  float M = rv[0]; __syncthreads();
  rv[tid] = psum[tid]; __syncthreads();
  for (int off = 128; off; off >>= 1) { if (tid < off) rv[tid] += rv[tid + off]; __syncthreads(); }
  float invS = 1.0f / rv[0]; __syncthreads();

  int base = blockIdx.x * CHUNK;
  for (int i = tid; i < CHUNK; i += 256) {
    int v = base + i;
    float p = -1.0f;
    if (v < V) {
      p = __expf(logits[v] - M) * invS * mask[v];
      mp_out[v] = p;
    }
    sv[i] = p;
  }
  __syncthreads();
  for (int r = 0; r < KTOP; r++) {
    float bv = -2.0f; int bp = 0;
    for (int i = tid; i < CHUNK; i += 256) if (sv[i] > bv) { bv = sv[i]; bp = i; }
    rv[tid] = bv; rp[tid] = bp; __syncthreads();
    for (int off = 128; off; off >>= 1) {
      if (tid < off && (rv[tid + off] > rv[tid] ||
                        (rv[tid + off] == rv[tid] && rp[tid + off] < rp[tid]))) {
        rv[tid] = rv[tid + off]; rp[tid] = rp[tid + off];
      }
      __syncthreads();
    }
    if (tid == 0) {
      cand_v[blockIdx.x * KTOP + r] = rv[0];
      cand_i[blockIdx.x * KTOP + r] = base + rp[0];
      sv[rp[0]] = -2.0f;
    }
    __syncthreads();
  }
}

// ---------------- merge 64*20 candidates -> global top-20 ----------------
__global__ __launch_bounds__(256)
void k_merge(const float* __restrict__ cand_v, const int* __restrict__ cand_i,
             int* __restrict__ top_i) {
  __shared__ float cv[NCAND];
  __shared__ int   ci[NCAND];
  __shared__ float rv[256];
  __shared__ int   rp[256];
  int tid = threadIdx.x;
  for (int i = tid; i < NCAND; i += 256) { cv[i] = cand_v[i]; ci[i] = cand_i[i]; }
  __syncthreads();
  for (int r = 0; r < KTOP; r++) {
    float bv = -3.0f; int bp = 0;
    for (int i = tid; i < NCAND; i += 256) if (cv[i] > bv) { bv = cv[i]; bp = i; }
    rv[tid] = bv; rp[tid] = bp; __syncthreads();
    for (int off = 128; off; off >>= 1) {
      if (tid < off && (rv[tid + off] > rv[tid] ||
                        (rv[tid + off] == rv[tid] && rp[tid + off] < rp[tid]))) {
        rv[tid] = rv[tid + off]; rp[tid] = rp[tid + off];
      }
      __syncthreads();
    }
    if (tid == 0) { top_i[r] = ci[rp[0]]; cv[rp[0]] = -3.0f; }
    __syncthreads();
  }
}

// ---------------- hit test vs ground truth, a_hat / feedback / mask update ----------------
__global__ void k_update(const int* __restrict__ x, const int* __restrict__ top_i,
                         float* __restrict__ mask, int* ah_p, float* fb_p,
                         float* outA, float* outF) {
  __shared__ int hit[KTOP];
  int tid = threadIdx.x;
  if (tid < KTOP) {
    int item = top_i[tid];
    int hh = 0;
    for (int j = 0; j < 200; j++) hh |= (x[j] == item + 1);
    hit[tid] = hh;
  }
  __syncthreads();
  if (tid == 0) {
    int card = 0, first = -1;
    for (int k = 0; k < KTOP; k++) { if (hit[k]) { card++; if (first < 0) first = k; } }
    int sel = (card > 0) ? top_i[first] : top_i[0];
    float fb = (card > 0) ? 1.0f : -1.0f;
    mask[sel] = 0.0f;
    *ah_p = sel; *fb_p = fb;
    *outA = (float)sel; *outF = fb;
  }
}

extern "C" void kernel_launch(void* const* d_in, const int* in_sizes, int n_in,
                              void* d_out, int out_size, void* d_ws, size_t ws_size,
                              hipStream_t stream) {
  (void)in_sizes; (void)n_in; (void)out_size;
  const int*   x     = (const int*)  d_in[0];
  const float* embed = (const float*)d_in[1];
  const float* Wih   = (const float*)d_in[2];
  const float* Whh   = (const float*)d_in[3];
  const float* bih   = (const float*)d_in[4];
  const float* bhh   = (const float*)d_in[5];
  const float* Wout  = (const float*)d_in[6];
  const float* bout  = (const float*)d_in[7];

  char* ws = (char*)d_ws;
  size_t o = 0;
  auto al = [&](size_t sz) { size_t r = o; o = (o + sz + 255) & ~(size_t)255; return r; };
  size_t O_MASK = al((size_t)V * 4);
  size_t O_LOG  = al((size_t)V * 4);
  size_t O_H    = al(H * 4);
  size_t O_C    = al(H * 4);
  size_t O_HB   = al(H * 2);
  size_t O_G    = al(G4 * 4);
  size_t O_PMAX = al(12500 * 4);
  size_t O_PSUM = al(256 * 4);
  size_t O_CV   = al(NCAND * 4);
  size_t O_CI   = al(NCAND * 4);
  size_t O_TI   = al(KTOP * 4);
  size_t O_SC   = al(64);
  size_t O_WB   = (o + 1023) & ~(size_t)1023;
  bool use_bf16 = ws_size >= O_WB + (size_t)V * H * 2;

  float* mask   = (float*)(ws + O_MASK);
  float* logits = (float*)(ws + O_LOG);
  float* hf     = (float*)(ws + O_H);
  float* cf     = (float*)(ws + O_C);
  bf16*  hb     = (bf16*) (ws + O_HB);
  float* gates  = (float*)(ws + O_G);
  float* pmax   = (float*)(ws + O_PMAX);
  float* psum   = (float*)(ws + O_PSUM);
  float* candv  = (float*)(ws + O_CV);
  int*   candi  = (int*)  (ws + O_CI);
  int*   topi   = (int*)  (ws + O_TI);
  int*   ah     = (int*)  (ws + O_SC);
  float* fb     = (float*)(ws + O_SC + 4);
  bf16*  Wb     = (bf16*) (ws + O_WB);

  float* out  = (float*)d_out;
  float* outA = out;                                   // a_hats [T]
  float* outF = out + T_STEPS;                         // feedbacks [T]
  float* outM = out + 2 * T_STEPS;                     // masked_probs [T*V]
  float* outH = outM + (size_t)T_STEPS * V;            // hidden_states [T*H]
  float* outC = outH + (size_t)T_STEPS * H;            // cell_states [T*H]

  k_init<<<(V + 255) / 256, 256, 0, stream>>>(mask, hf, cf, ah, fb);
  if (use_bf16)
    k_convert<<<2048, 256, 0, stream>>>(Wout, Wb, (long)V * H);

  int nb = use_bf16 ? 1563 : 12500;
  for (int t = 0; t < T_STEPS; t++) {
    k_gates<<<512, 256, 0, stream>>>(Wih, Whh, bih, bhh, embed, hf, ah, fb, gates);
    k_lstm<<<1, 1024, 0, stream>>>(gates, hf, cf, hb, outH + (size_t)t * H, outC + (size_t)t * H);
    if (use_bf16)
      k_logits_wmma<<<1563, 128, 0, stream>>>(Wb, hb, bout, logits, pmax);
    else
      k_logits_f32<<<12500, 256, 0, stream>>>(Wout, bout, hf, logits, pmax);
    k_expsum<<<256, 256, 0, stream>>>(logits, pmax, nb, psum);
    k_normtopk<<<TOPK_BLOCKS, 256, 0, stream>>>(logits, pmax, nb, psum, mask,
                                                outM + (size_t)t * V, candv, candi);
    k_merge<<<1, 256, 0, stream>>>(candv, candi, topi);
    k_update<<<1, 256, 0, stream>>>(x, topi, mask, ah, fb,
                                    outA + t, outF + t);
  }
}
